// MoleculeGNN_63290638074075
// MI455X (gfx1250) — matrix operand order repaired and checked by reading, not compiled, not run on hardware
//
#include <hip/hip_runtime.h>

// ---------------------------------------------------------------------------
// MoleculeGNN for MI455X (gfx1250, wave32, WMMA).
// Fused MLP: both K=128 GEMMs of each GIN layer in one kernel, computed in
// transposed orientation (weights = WMMA A operand from LDS, activations =
// B operand). Intermediate 16x128 tile lives in per-wave LDS; all epilogues
// are b128-vectorized. h stays fp32 (atomics + BN fidelity).
// ---------------------------------------------------------------------------

#define N_NODES  200000
#define N_EDGES  600000
#define N_GRAPHS 10000
#define HDIM     128
#define BN_EPS   1e-5f

typedef __attribute__((ext_vector_type(16))) _Float16 v16h;
typedef __attribute__((ext_vector_type(8)))  _Float16 v8h;
typedef __attribute__((ext_vector_type(8)))  float    v8f;

// ------------------------------ utility kernels ----------------------------

__global__ void zero_f32(float* __restrict__ p, long long n) {
  long long i = (long long)blockIdx.x * blockDim.x + threadIdx.x;
  long long s = (long long)gridDim.x * blockDim.x;
  for (; i < n; i += s) p[i] = 0.0f;
}

// msg = relu(h[src]); aggr[dst] += msg  — generic (used for the d=5 layer 0)
template<int D>
__global__ void scatter_relu(const float* __restrict__ h, const int* __restrict__ src,
                             const int* __restrict__ dst, float* __restrict__ aggr, int E) {
  long long total = (long long)E * D;
  long long i = (long long)blockIdx.x * blockDim.x + threadIdx.x;
  long long s = (long long)gridDim.x * blockDim.x;
  for (; i < total; i += s) {
    int e = (int)(i / D);
    int f = (int)(i - (long long)e * D);
    float v = h[(long long)src[e] * D + f];
    if (v > 0.0f) atomicAdd(&aggr[(long long)dst[e] * D + f], v);
  }
}

// d=128 scatter, float4-vectorized gather: 1x global_load_b128 + 4 atomics
__global__ void scatter_relu128_v4(const float* __restrict__ h, const int* __restrict__ src,
                                   const int* __restrict__ dst, float* __restrict__ aggr, int E) {
  long long total = (long long)E * 32;       // 128 floats = 32 float4 per edge
  long long i = (long long)blockIdx.x * blockDim.x + threadIdx.x;
  long long s = (long long)gridDim.x * blockDim.x;
  for (; i < total; i += s) {
    int e  = (int)(i >> 5);
    int f4 = (int)(i & 31);
    const float4 v = *(const float4*)(h + (long long)src[e] * HDIM + f4 * 4);
    float* d = aggr + (long long)dst[e] * HDIM + f4 * 4;
    if (v.x > 0.0f) atomicAdd(d + 0, v.x);
    if (v.y > 0.0f) atomicAdd(d + 1, v.y);
    if (v.z > 0.0f) atomicAdd(d + 2, v.z);
    if (v.w > 0.0f) atomicAdd(d + 3, v.w);
  }
}

__global__ void add_f32(const float* __restrict__ a, const float* __restrict__ b,
                        float* __restrict__ c, long long n) {
  long long i = (long long)blockIdx.x * blockDim.x + threadIdx.x;
  long long s = (long long)gridDim.x * blockDim.x;
  for (; i < n; i += s) c[i] = a[i] + b[i];
}

// z16 = (f16)(h + aggr)  -> WMMA B operand
__global__ void add_convert_f16(const float* __restrict__ a, const float* __restrict__ b,
                                _Float16* __restrict__ c, long long n) {
  long long i = (long long)blockIdx.x * blockDim.x + threadIdx.x;
  long long s = (long long)gridDim.x * blockDim.x;
  for (; i < n; i += s) c[i] = (_Float16)(a[i] + b[i]);
}

// W[K][Nout] fp32 -> Wt[Nout][K] f16 (A operand: lane reads its feature row)
__global__ void transpose_w(const float* __restrict__ W, _Float16* __restrict__ Wt,
                            int K, int Nout) {
  long long total = (long long)K * Nout;
  long long i = (long long)blockIdx.x * blockDim.x + threadIdx.x;
  long long s = (long long)gridDim.x * blockDim.x;
  for (; i < total; i += s) {
    int k = (int)(i / Nout);
    int n = (int)(i - (long long)k * Nout);
    Wt[(long long)n * K + k] = (_Float16)W[i];
  }
}

// layer 0 first GEMM: [N,5] x [5,128]  (K=5 too small for WMMA; scalar is cheap)
__global__ void gemm_in5(const float* __restrict__ z, const float* __restrict__ W1,
                         const float* __restrict__ b1, _Float16* __restrict__ out, int M) {
  int i = blockIdx.x * blockDim.x + threadIdx.x;
  if (i >= M * HDIM) return;
  int m = i >> 7, c = i & 127;
  float acc = b1[c];
#pragma unroll
  for (int k = 0; k < 5; ++k) acc += z[m * 5 + k] * W1[k * HDIM + c];
  out[i] = (_Float16)fmaxf(acc, 0.0f);
}

// ------------------------------ fused WMMA MLP -----------------------------
// Transposed orientation: D_tile[f][m] = sum_k Wt[f][k] * In[m][k].
//   A operand (16xK tile of weights)  : from LDS, contiguous K halves
//   B operand (Kx16 tile, activations): lane = node row, contiguous K halves
//   D tile: lane l -> node m = l&15, feature f = r + 8*(l>>4)  (vector epilogues)
// DO1: GEMM1 (+b1, ReLU) -> ytile in per-wave LDS (or f16 global if !DO2)
// DO2: GEMM2 (+b2 [, BN], ReLU) -> fp32 global
template<int NT1, int NT2, bool DO1, bool DO2, bool BN2>
__launch_bounds__(256)
__global__ void wmma_mlp(const _Float16* __restrict__ In,
                         const _Float16* __restrict__ W1t, const float* __restrict__ b1,
                         const _Float16* __restrict__ W2t, const float* __restrict__ b2,
                         const float* __restrict__ gamma, const float* __restrict__ beta,
                         const float* __restrict__ mu,    const float* __restrict__ var,
                         _Float16* __restrict__ out16, float* __restrict__ out32, int M) {
  constexpr int K  = 128;
  constexpr int YS = 136;                      // ytile row stride: 272B -> bank step 4
  extern __shared__ _Float16 smem[];
  _Float16* w1s = smem;
  _Float16* w2s = smem + (DO1 ? NT1 * 16 * K : 0);
  _Float16* yts = w2s + (DO2 ? NT2 * 16 * K : 0);

  // stage weights in LDS (uint4 = 8 halves per transfer)
  if (DO1) {
    const uint4* s = (const uint4*)W1t; uint4* d = (uint4*)w1s;
    for (int i = threadIdx.x; i < NT1 * 16 * K / 8; i += blockDim.x) d[i] = s[i];
  }
  if (DO2) {
    const uint4* s = (const uint4*)W2t; uint4* d = (uint4*)w2s;
    for (int i = threadIdx.x; i < NT2 * 16 * K / 8; i += blockDim.x) d[i] = s[i];
  }
  __syncthreads();

  const int wave = threadIdx.x >> 5;
  const int lane = threadIdx.x & 31;
  int m_base = blockIdx.x * 128 + wave * 16;
  if (m_base + 16 > M) m_base = M - 16;        // tail: duplicate rows, identical writes
  const int hsel = lane >> 4;                  // 0|1
  const int l15  = lane & 15;

  // B fragments: lane = node row (m_base+l15), contiguous 16 K-halves @ hsel*16
  v16h bfrag[4];
  {
    const _Float16* inrow = In + (long long)(m_base + l15) * K + hsel * 16;
    __builtin_prefetch(inrow + (long long)128 * K, 0, 1);  // next block's stripe
#pragma unroll
    for (int kt = 0; kt < 4; ++kt) bfrag[kt] = *(const v16h*)(inrow + kt * 32);
  }

  // helper: A fragment from LDS (contiguous layout: k = kt*32 + hsel*16 + s)
  auto loadA = [&](const _Float16* ws, int nt, int kt) -> v16h {
    const _Float16* arow = ws + (nt * 16 + l15) * K + kt * 32 + hsel * 16;
    return *(const v16h*)(arow);
  };

  _Float16* ywave = yts + wave * 16 * YS;

  if (DO1) {
#pragma unroll
    for (int nt = 0; nt < NT1; ++nt) {
      v16h af[4];
#pragma unroll
      for (int kt = 0; kt < 4; ++kt) af[kt] = loadA(w1s, nt, kt);  // load clause first
      v8f acc = {};
#pragma unroll
      for (int kt = 0; kt < 4; ++kt)                                // back-to-back WMMA
        acc = __builtin_amdgcn_wmma_f32_16x16x32_f16(
            false, af[kt], false, bfrag[kt], (short)0, acc, false, false);
      // epilogue 1: f = nt*16 + hsel*8 + r  (8 consecutive features per lane)
      const int off = nt * 16 + hsel * 8;
      float bb[8];
      *(float4*)&bb[0] = *(const float4*)(b1 + off);
      *(float4*)&bb[4] = *(const float4*)(b1 + off + 4);
      v8h pk;
#pragma unroll
      for (int r = 0; r < 8; ++r) pk[r] = (_Float16)fmaxf(acc[r] + bb[r], 0.0f);
      if (DO2)  *(v8h*)(ywave + l15 * YS + off) = pk;                        // ds_store_b128
      else      *(v8h*)(out16 + (long long)(m_base + l15) * (NT1 * 16) + off) = pk;
    }
  }

  if (DO1 && DO2) {
    // RAW through LDS within the same wave: DS ops are in-order per wave;
    // fence explicitly and keep the compiler from reordering.
    asm volatile("s_wait_dscnt 0x0" ::: "memory");
    const _Float16* yrow = ywave + l15 * YS + hsel * 16;
#pragma unroll
    for (int kt = 0; kt < 4; ++kt) bfrag[kt] = *(const v16h*)(yrow + kt * 32);
  }

  if (DO2) {
#pragma unroll
    for (int nt = 0; nt < NT2; ++nt) {
      v16h af[4];
#pragma unroll
      for (int kt = 0; kt < 4; ++kt) af[kt] = loadA(w2s, nt, kt);
      v8f acc = {};
#pragma unroll
      for (int kt = 0; kt < 4; ++kt)
        acc = __builtin_amdgcn_wmma_f32_16x16x32_f16(
            false, af[kt], false, bfrag[kt], (short)0, acc, false, false);
      // epilogue 2: c = nt*16 + hsel*8 + r ; out row = node (m_base+l15)
      const int off = nt * 16 + hsel * 8;
      float bb[8], sc[8], sh[8];
      *(float4*)&bb[0] = *(const float4*)(b2 + off);
      *(float4*)&bb[4] = *(const float4*)(b2 + off + 4);
      if (BN2) {
        float gv[8], bv[8], mv[8], vv[8];
        *(float4*)&gv[0] = *(const float4*)(gamma + off);
        *(float4*)&gv[4] = *(const float4*)(gamma + off + 4);
        *(float4*)&bv[0] = *(const float4*)(beta + off);
        *(float4*)&bv[4] = *(const float4*)(beta + off + 4);
        *(float4*)&mv[0] = *(const float4*)(mu + off);
        *(float4*)&mv[4] = *(const float4*)(mu + off + 4);
        *(float4*)&vv[0] = *(const float4*)(var + off);
        *(float4*)&vv[4] = *(const float4*)(var + off + 4);
#pragma unroll
        for (int r = 0; r < 8; ++r) {
          sc[r] = rsqrtf(vv[r] + BN_EPS) * gv[r];
          sh[r] = bv[r] - mv[r] * sc[r];
        }
      }
      float o[8];
#pragma unroll
      for (int r = 0; r < 8; ++r) {
        float v = acc[r] + bb[r];
        if (BN2) v = v * sc[r] + sh[r];
        o[r] = fmaxf(v, 0.0f);
      }
      float* orow = out32 + (long long)(m_base + l15) * (NT2 * 16) + off;
      *(float4*)(orow)     = make_float4(o[0], o[1], o[2], o[3]);
      *(float4*)(orow + 4) = make_float4(o[4], o[5], o[6], o[7]);
    }
  }
}

// ------------------------------ pooling + head -----------------------------

__global__ void pool_scatter(const float* __restrict__ h, const int* __restrict__ batch,
                             float* __restrict__ psum, float* __restrict__ cnt, int Nn) {
  long long total = (long long)Nn * 32;      // float4 granularity
  long long i = (long long)blockIdx.x * blockDim.x + threadIdx.x;
  long long s = (long long)gridDim.x * blockDim.x;
  for (; i < total; i += s) {
    int n  = (int)(i >> 5);
    int f4 = (int)(i & 31);
    int g  = batch[n];
    const float4 v = *(const float4*)(h + (long long)n * HDIM + f4 * 4);
    float* d = psum + (long long)g * HDIM + f4 * 4;
    atomicAdd(d + 0, v.x);
    atomicAdd(d + 1, v.y);
    atomicAdd(d + 2, v.z);
    atomicAdd(d + 3, v.w);
    if (f4 == 0) atomicAdd(&cnt[g], 1.0f);
  }
}

__global__ void pool_finalize(const float* __restrict__ psum, const float* __restrict__ cnt,
                              _Float16* __restrict__ p16, int Gn) {
  long long total = (long long)Gn * HDIM;
  long long i = (long long)blockIdx.x * blockDim.x + threadIdx.x;
  long long s = (long long)gridDim.x * blockDim.x;
  for (; i < total; i += s) {
    int g = (int)(i >> 7);
    p16[i] = (_Float16)(psum[i] / fmaxf(cnt[g], 1.0f));
  }
}

__global__ void lin2_kernel(const _Float16* __restrict__ y, const float* __restrict__ w,
                            const float* __restrict__ b, float* __restrict__ out, int Gn) {
  int g = blockIdx.x * blockDim.x + threadIdx.x;
  if (g >= Gn) return;
  float acc = b[0];
#pragma unroll
  for (int k = 0; k < 64; ++k) acc += (float)y[(long long)g * 64 + k] * w[k];
  out[g] = acc;
}

// ------------------------------ orchestration ------------------------------

extern "C" void kernel_launch(void* const* d_in, const int* in_sizes, int n_in,
                              void* d_out, int out_size, void* d_ws, size_t ws_size,
                              hipStream_t stream) {
  (void)in_sizes; (void)n_in; (void)out_size; (void)ws_size;

  const float* x     = (const float*)d_in[0];
  const int*   ei    = (const int*)d_in[1];
  const int*   src   = ei;
  const int*   dst   = ei + N_EDGES;
  const int*   batch = (const int*)d_in[2];
  auto LP = [&](int layer, int j) { return (const float*)d_in[3 + layer * 8 + j]; };
  const float* lin1_w = (const float*)d_in[27];
  const float* lin1_b = (const float*)d_in[28];
  const float* lin2_w = (const float*)d_in[29];
  const float* lin2_b = (const float*)d_in[30];

  // workspace carve (h: 102 MB, fits the 192 MB L2 -> gather/scatter L2-resident)
  char* wp = (char*)d_ws;
  auto carve = [&](size_t bytes) -> void* {
    void* r = (void*)wp;
    wp += (bytes + 255) & ~(size_t)255;
    return r;
  };
  float*    h     = (float*)carve((size_t)N_NODES * HDIM * 4);
  float*    aggr  = (float*)carve((size_t)N_NODES * HDIM * 4);
  _Float16* z16   = (_Float16*)carve((size_t)N_NODES * HDIM * 2);
  _Float16* y16   = (_Float16*)carve((size_t)N_NODES * HDIM * 2);
  float*    z5    = (float*)carve((size_t)N_NODES * 5 * 4);
  float*    psum  = (float*)carve((size_t)N_GRAPHS * HDIM * 4);
  float*    cnt   = (float*)carve((size_t)N_GRAPHS * 4);
  _Float16* p16   = (_Float16*)carve((size_t)N_GRAPHS * HDIM * 2);
  _Float16* y64   = (_Float16*)carve((size_t)N_GRAPHS * 64 * 2);
  _Float16* W2t0  = (_Float16*)carve(128 * 128 * 2);
  _Float16* W1t1  = (_Float16*)carve(128 * 128 * 2);
  _Float16* W2t1  = (_Float16*)carve(128 * 128 * 2);
  _Float16* W1t2  = (_Float16*)carve(128 * 128 * 2);
  _Float16* W2t2  = (_Float16*)carve(128 * 128 * 2);
  _Float16* lin1t = (_Float16*)carve(64 * 128 * 2);

  const int TB = 256;
  const int tg128 = (128 * 128 + TB - 1) / TB;
  transpose_w<<<tg128, TB, 0, stream>>>(LP(0, 2), W2t0, 128, 128);
  transpose_w<<<tg128, TB, 0, stream>>>(LP(1, 0), W1t1, 128, 128);
  transpose_w<<<tg128, TB, 0, stream>>>(LP(1, 2), W2t1, 128, 128);
  transpose_w<<<tg128, TB, 0, stream>>>(LP(2, 0), W1t2, 128, 128);
  transpose_w<<<tg128, TB, 0, stream>>>(LP(2, 2), W2t2, 128, 128);
  transpose_w<<<(128 * 64 + TB - 1) / TB, TB, 0, stream>>>(lin1_w, lin1t, 128, 64);

  // dynamic-LDS sizes
  const int SM_FULL = (8 * 16 * 128 + 8 * 16 * 128 + 8 * 16 * 136) * 2;  // 100352 B
  const int SM_G2   = (8 * 16 * 128) * 2;                                // 32768 B
  const int SM_G1   = (4 * 16 * 128) * 2;                                // 16384 B
  hipFuncSetAttribute((const void*)wmma_mlp<8, 8, true, true, true>,
                      hipFuncAttributeMaxDynamicSharedMemorySize, SM_FULL);

  const int gblocks = (N_NODES + 127) / 128;   // 1563 (M % 16 == 0)

  // ---- layer 0 (fin = 5): scalar GEMM1, WMMA GEMM2+BN ----
  zero_f32<<<2048, TB, 0, stream>>>(aggr, (long long)N_NODES * 5);
  scatter_relu<5><<<4096, TB, 0, stream>>>(x, src, dst, aggr, N_EDGES);
  add_f32<<<2048, TB, 0, stream>>>(x, aggr, z5, (long long)N_NODES * 5);
  gemm_in5<<<(N_NODES * HDIM + TB - 1) / TB, TB, 0, stream>>>(z5, LP(0, 0), LP(0, 1), y16, N_NODES);
  wmma_mlp<1, 8, false, true, true><<<gblocks, TB, SM_G2, stream>>>(
      y16, nullptr, nullptr, W2t0, LP(0, 3), LP(0, 4), LP(0, 5), LP(0, 6), LP(0, 7),
      nullptr, h, N_NODES);

  // ---- layers 1, 2 (fin = 128): fully fused MLP ----
  _Float16* W1t[2] = {W1t1, W1t2};
  _Float16* W2t[2] = {W2t1, W2t2};
  for (int l = 1; l < 3; ++l) {
    zero_f32<<<4096, TB, 0, stream>>>(aggr, (long long)N_NODES * HDIM);
    scatter_relu128_v4<<<4096, TB, 0, stream>>>(h, src, dst, aggr, N_EDGES);
    add_convert_f16<<<4096, TB, 0, stream>>>(h, aggr, z16, (long long)N_NODES * HDIM);
    wmma_mlp<8, 8, true, true, true><<<gblocks, TB, SM_FULL, stream>>>(
        z16, W1t[l - 1], LP(l, 1), W2t[l - 1], LP(l, 3), LP(l, 4), LP(l, 5), LP(l, 6), LP(l, 7),
        nullptr, h, N_NODES);
  }

  // ---- global mean pool + head ----
  zero_f32<<<512, TB, 0, stream>>>(psum, (long long)N_GRAPHS * HDIM);
  zero_f32<<<64, TB, 0, stream>>>(cnt, N_GRAPHS);
  pool_scatter<<<2048, TB, 0, stream>>>(h, batch, psum, cnt, N_NODES);
  pool_finalize<<<512, TB, 0, stream>>>(psum, cnt, p16, N_GRAPHS);
  wmma_mlp<4, 1, true, false, false><<<(N_GRAPHS + 127) / 128, TB, SM_G1, stream>>>(
      p16, lin1t, lin1_b, nullptr, nullptr, nullptr, nullptr, nullptr, nullptr,
      y64, nullptr, N_GRAPHS);
  lin2_kernel<<<(N_GRAPHS + TB - 1) / TB, TB, 0, stream>>>(y64, lin2_w, lin2_b, (float*)d_out, N_GRAPHS);
}
